// SGC_lr_84954453114997
// MI455X (gfx1250) — matrix operand order repaired and checked by reading, not compiled, not run on hardware
//
#include <hip/hip_runtime.h>
#include <hip/hip_bf16.h>

// ---------------------------------------------------------------------------
// SGC on MI455X (gfx1250):
//   h0 = LayerNorm(GELU(x @ W1 + b1))
//   h  = Â^2 h0   (sym-normalized adjacency with self loops, via f32 atomics)
//   t  = LayerNorm(GELU(h @ Wc + bc))
//   out= t @ W2 + b2
// GEMMs: v_wmma_f32_16x16x32_bf16, A tile staged once per block in LDS (bf16).
// ---------------------------------------------------------------------------

typedef __attribute__((ext_vector_type(16))) __bf16 bf16x16;
typedef __attribute__((ext_vector_type(8)))  __bf16 bf16x8;
typedef __attribute__((ext_vector_type(8)))  float  f32x8;

__device__ __forceinline__ float gelu_exact(float x) {
    return 0.5f * x * (1.0f + erff(x * 0.70710678118654752440f));
}

// ---------------------------------------------------------------------------
// Weight convert + transpose: W [K x NOUT] f32 (row-major) -> Wt [NOUT x K] bf16
// ---------------------------------------------------------------------------
__global__ void wtrans_bf16_kernel(const float* __restrict__ W,
                                   __bf16* __restrict__ Wt,
                                   int K, int NOUT) {
    int idx = blockIdx.x * blockDim.x + threadIdx.x;
    if (idx >= K * NOUT) return;
    int k = idx / NOUT;
    int n = idx - k * NOUT;
    Wt[(size_t)n * K + k] = (__bf16)W[idx];
}

// ---------------------------------------------------------------------------
// Degree / normalization
// ---------------------------------------------------------------------------
__global__ void deg_init_kernel(float* __restrict__ deg, int n) {
    int i = blockIdx.x * blockDim.x + threadIdx.x;
    if (i < n) deg[i] = 1.0f;   // self loop contributes 1
}

__global__ void deg_edges_kernel(const int* __restrict__ dst,
                                 float* __restrict__ deg, int e) {
    int i = blockIdx.x * blockDim.x + threadIdx.x;
    if (i < e) unsafeAtomicAdd(&deg[dst[i]], 1.0f);
}

__global__ void deg_to_dinv_kernel(float* __restrict__ deg, int n) {
    int i = blockIdx.x * blockDim.x + threadIdx.x;
    if (i < n) {
        float d = deg[i];
        deg[i] = (d > 0.0f) ? rsqrtf(d) : 0.0f;
    }
}

// ---------------------------------------------------------------------------
// Propagation hop: out = D^-1/2 (A + I) D^-1/2 * h   (128 channels)
// ---------------------------------------------------------------------------
__global__ void prop_init_kernel(const float* __restrict__ h,
                                 const float* __restrict__ dinv,
                                 float* __restrict__ out, int total) {
    int i = blockIdx.x * blockDim.x + threadIdx.x;
    if (i >= total) return;
    int row = i >> 7;          // 128 channels
    float d = dinv[row];
    out[i] = d * d * h[i];
}

// One wave (32 lanes) per edge: e is wave-uniform (scalar loads for src/dst),
// each lane moves 4 channels: coalesced 512 B gather + 4 fp32 atomics.
__global__ void prop_edges_kernel(const int* __restrict__ src,
                                  const int* __restrict__ dst,
                                  const float* __restrict__ dinv,
                                  const float* __restrict__ hin,
                                  float* __restrict__ hout, int E) {
    int t = blockIdx.x * blockDim.x + threadIdx.x;
    int e = t >> 5;
    if (e >= E) return;
    int lane = t & 31;
    int s = src[e];
    int d = dst[e];
    float coef = dinv[s] * dinv[d];
    const float4 v = *((const float4*)(hin + (size_t)s * 128) + lane);
    float* op = hout + (size_t)d * 128 + lane * 4;
    unsafeAtomicAdd(op + 0, coef * v.x);
    unsafeAtomicAdd(op + 1, coef * v.y);
    unsafeAtomicAdd(op + 2, coef * v.z);
    unsafeAtomicAdd(op + 3, coef * v.w);
}

// ---------------------------------------------------------------------------
// WMMA GEMM: per block, out[16 x NOUT] = A[16 x K] @ Bt^T + bias,
// optional fused GELU + LayerNorm (POST; requires NOUT==128, 256 threads).
//   A  : f32 row-major [N x K]  (staged once per block into LDS as bf16)
//   Bt : bf16 [NOUT x K]        (transposed weights; L2-resident)
// Wave w computes the 16x16 tile at cols [16w, 16w+16). Requires N % 16 == 0.
// ---------------------------------------------------------------------------
template <int K, int NOUT, bool POST>
__global__ void gemm_bf16_wmma_kernel(const float* __restrict__ A,
                                      const __bf16* __restrict__ Bt,
                                      const float* __restrict__ bias,
                                      const float* __restrict__ gamma,
                                      const float* __restrict__ beta,
                                      float* __restrict__ out) {
    static_assert(!POST || NOUT == 128, "POST path assumes NOUT=128/256 threads");
    constexpr int LDK = K + 8;                     // padded bf16 row stride
    __shared__ __bf16 lA[16 * LDK];
    __shared__ float  tile[16][NOUT + 8];
    __shared__ float  s_mu[16];
    __shared__ float  s_rs[16];

    const int r0   = blockIdx.x * 16;
    const int wave = threadIdx.x >> 5;
    const int lane = threadIdx.x & 31;
    const int half = lane >> 4;      // 0: lanes 0-15, 1: lanes 16-31
    const int m    = lane & 15;
    const int col  = wave * 16 + m;

    // --- Stage A tile (16 x K) f32 -> bf16 into LDS, one pass per block ----
    {
        constexpr int CHUNKS = 16 * (K / 8);       // 8 elements per chunk
        for (int c = threadIdx.x; c < CHUNKS; c += blockDim.x) {
            int r  = c / (K / 8);
            int kp = (c - r * (K / 8)) * 8;
            const float* sp = A + (size_t)(r0 + r) * K + kp;
            float4 u = *(const float4*)(sp);
            float4 v = *(const float4*)(sp + 4);
            bf16x8 w;
            w[0] = (__bf16)u.x; w[1] = (__bf16)u.y;
            w[2] = (__bf16)u.z; w[3] = (__bf16)u.w;
            w[4] = (__bf16)v.x; w[5] = (__bf16)v.y;
            w[6] = (__bf16)v.z; w[7] = (__bf16)v.w;
            *(bf16x8*)(&lA[r * LDK + kp]) = w;     // ds_store_b128
        }
    }
    __syncthreads();

    // A fragment (16-bit A 16x32): half 0 -> K chunks [0,8)+[16,24),
    //                              half 1 -> K chunks [8,16)+[24,32)
    const int koffA = half * 8;
    // B fragment (16-bit B 32x16): half 0 -> K chunks [0,8)+[16,24),
    //                              half 1 -> K chunks [8,16)+[24,32)
    const int koffB = half * 8;

    const __bf16* lap = &lA[m * LDK];
    const __bf16* bp  = Bt + (size_t)col * K;

    f32x8 acc = {};
    for (int kk = 0; kk < K; kk += 32) {
        const __bf16* a0 = lap + kk + koffA;
        bf16x8 alo = *(const bf16x8*)(a0);         // ds_load_b128
        bf16x8 ahi = *(const bf16x8*)(a0 + 16);    // ds_load_b128
        bf16x16 a = __builtin_shufflevector(alo, ahi,
            0,1,2,3,4,5,6,7,8,9,10,11,12,13,14,15);

        const __bf16* b0 = bp + kk + koffB;
        bf16x8 blo = *(const bf16x8*)(b0);         // global_load_b128
        bf16x8 bhi = *(const bf16x8*)(b0 + 16);
        bf16x16 b = __builtin_shufflevector(blo, bhi,
            0,1,2,3,4,5,6,7,8,9,10,11,12,13,14,15);

        acc = __builtin_amdgcn_wmma_f32_16x16x32_bf16(
            /*neg_a=*/false, a, /*neg_b=*/false, b,
            /*c_mod=*/(short)0, acc, /*reuse_a=*/false, /*reuse_b=*/false);
    }

    // C/D layout: VGPR i -> row (i + 8*half), col = lane&15 (within tile)
    if constexpr (POST) {
#pragma unroll
        for (int i = 0; i < 8; ++i) {
            int r = i + half * 8;
            tile[r][col] = gelu_exact(acc[i] + bias[col]);
        }
        __syncthreads();
        // LayerNorm stats: 16 lanes per row + shfl_xor tree (stays in 16-group)
        {
            int rr = threadIdx.x >> 4;     // row 0..15
            int jj = threadIdx.x & 15;
            float s = 0.0f, s2 = 0.0f;
            for (int c = jj; c < NOUT; c += 16) {
                float v = tile[rr][c];
                s += v; s2 += v * v;
            }
#pragma unroll
            for (int o = 8; o > 0; o >>= 1) {
                s  += __shfl_xor(s,  o, 32);
                s2 += __shfl_xor(s2, o, 32);
            }
            if (jj == 0) {
                float mu  = s * (1.0f / NOUT);
                float var = s2 * (1.0f / NOUT) - mu * mu;
                s_mu[rr] = mu;
                s_rs[rr] = rsqrtf(var + 1e-5f);
            }
        }
        __syncthreads();
        const int total = 16 * NOUT;
        for (int idx = threadIdx.x; idx < total; idx += blockDim.x) {
            int r = idx / NOUT;
            int c = idx - r * NOUT;
            float v = (tile[r][c] - s_mu[r]) * s_rs[r];
            out[(size_t)(r0 + r) * NOUT + c] = v * gamma[c] + beta[c];
        }
    } else {
#pragma unroll
        for (int i = 0; i < 8; ++i) {
            int r = i + half * 8;
            out[(size_t)(r0 + r) * NOUT + col] = acc[i] + bias[col];
        }
    }
}

// ---------------------------------------------------------------------------
// Launch
// ---------------------------------------------------------------------------
extern "C" void kernel_launch(void* const* d_in, const int* in_sizes, int n_in,
                              void* d_out, int out_size, void* d_ws, size_t ws_size,
                              hipStream_t stream) {
    (void)n_in; (void)out_size; (void)ws_size;

    const float* x   = (const float*)d_in[0];
    const int*   ei  = (const int*)  d_in[1];
    const float* W1  = (const float*)d_in[2];
    const float* b1  = (const float*)d_in[3];
    const float* g1  = (const float*)d_in[4];
    const float* be1 = (const float*)d_in[5];
    const float* Wc  = (const float*)d_in[6];
    const float* bc  = (const float*)d_in[7];
    const float* g2  = (const float*)d_in[8];
    const float* be2 = (const float*)d_in[9];
    const float* W2  = (const float*)d_in[10];
    const float* b2  = (const float*)d_in[11];
    float* out = (float*)d_out;

    const int IN_CH = 512, HID = 128, OUT_CH = 64;
    const int N = in_sizes[0] / IN_CH;   // 100000
    const int E = in_sizes[1] / 2;       // 1600000
    const int* e_src = ei;
    const int* e_dst = ei + E;

    // Workspace carve-up (everything is fully written before being read).
    char* ws = (char*)d_ws;
    auto align_up = [](size_t v) { return (v + 255) & ~(size_t)255; };
    size_t off = 0;
    __bf16* W1t = (__bf16*)(ws + off); off = align_up(off + (size_t)IN_CH * HID * 2);
    __bf16* Wct = (__bf16*)(ws + off); off = align_up(off + (size_t)HID * HID * 2);
    __bf16* W2t = (__bf16*)(ws + off); off = align_up(off + (size_t)HID * OUT_CH * 2);
    float* dinv = (float*)(ws + off);  off = align_up(off + (size_t)N * 4);
    float* h0   = (float*)(ws + off);  off = align_up(off + (size_t)N * HID * 4);
    float* hA   = (float*)(ws + off);  off = align_up(off + (size_t)N * HID * 4);

    const int T = 256;

    // 1) bf16-transpose all weights
    wtrans_bf16_kernel<<<(IN_CH * HID + T - 1) / T, T, 0, stream>>>(W1, W1t, IN_CH, HID);
    wtrans_bf16_kernel<<<(HID * HID + T - 1) / T, T, 0, stream>>>(Wc, Wct, HID, HID);
    wtrans_bf16_kernel<<<(HID * OUT_CH + T - 1) / T, T, 0, stream>>>(W2, W2t, HID, OUT_CH);

    // 2) symmetric normalization coefficients
    deg_init_kernel<<<(N + T - 1) / T, T, 0, stream>>>(dinv, N);
    deg_edges_kernel<<<(E + T - 1) / T, T, 0, stream>>>(e_dst, dinv, E);
    deg_to_dinv_kernel<<<(N + T - 1) / T, T, 0, stream>>>(dinv, N);

    // 3) h0 = LN(GELU(x @ W1 + b1))       [WMMA bf16, K=512]
    gemm_bf16_wmma_kernel<512, 128, true>
        <<<N / 16, 256, 0, stream>>>(x, W1t, b1, g1, be1, h0);

    // 4) two propagation hops: h0 -> hA -> h0
    const int totalF = N * HID;
    const int edgeT  = (E * 32 + T - 1) / T;
    prop_init_kernel<<<(totalF + T - 1) / T, T, 0, stream>>>(h0, dinv, hA, totalF);
    prop_edges_kernel<<<edgeT, T, 0, stream>>>(e_src, e_dst, dinv, h0, hA, E);
    prop_init_kernel<<<(totalF + T - 1) / T, T, 0, stream>>>(hA, dinv, h0, totalF);
    prop_edges_kernel<<<edgeT, T, 0, stream>>>(e_src, e_dst, dinv, hA, h0, E);

    // 5) t = LN(GELU(h @ Wc + bc))        [WMMA bf16, K=128] -> hA
    gemm_bf16_wmma_kernel<128, 128, true>
        <<<N / 16, 256, 0, stream>>>(h0, Wct, bc, g2, be2, hA);

    // 6) out = t @ W2 + b2                [WMMA bf16, K=128, NOUT=64]
    gemm_bf16_wmma_kernel<128, 64, false>
        <<<N / 16, 128, 0, stream>>>(hA, W2t, b2, nullptr, nullptr, out);
}
